// NeuralCDE_5755256176726
// MI455X (gfx1250) — compile-verified, matrix operand order checked by reading, not run on hardware
//
#include <hip/hip_runtime.h>
#include <math.h>

#define BB 64
#define TT 512
#define HH 128
#define WW 128
#define DD 32
#define MT 16           // batch rows per workgroup (WMMA M)
#define NWAVE 16
#define NTHR (NWAVE * 32)

typedef __attribute__((ext_vector_type(16))) _Float16     v16h;
typedef __attribute__((ext_vector_type(8)))  float        v8f;
typedef __attribute__((ext_vector_type(4)))  unsigned int v4u;
typedef __attribute__((ext_vector_type(8)))  int          v8i;
typedef __attribute__((ext_vector_type(4)))  int          v4i;

// ---------------- f32 -> f16 weight conversion ----------------
__global__ void cvt_f32_f16(const float* __restrict__ src, _Float16* __restrict__ dst, int n) {
    int i = blockIdx.x * blockDim.x + threadIdx.x;
    if (i < n) dst[i] = (_Float16)src[i];
}

// ---------------- WMMA helpers ----------------
// Load a 16x32 chunk in the CDNA5 16-bit A-matrix layout from a row-major
// f16 matrix (row stride = `stride` halves). Works for LDS or global pointers.
// lane<16:  row=lane,    K = {0..7} U {16..23}
// lane>=16: row=lane-16, K = {8..15} U {24..31}
__device__ __forceinline__ v16h load_frag(const _Float16* p0, int stride) {
    int lane = threadIdx.x & 31;
    const _Float16* p = p0 + (lane & 15) * stride + ((lane & 16) ? 8 : 0);
    v16h r;
#pragma unroll
    for (int i = 0; i < 8; ++i) { r[i] = p[i]; r[8 + i] = p[16 + i]; }
    return r;
}

// 16x16 output tile: Out = A(16xK) * W^T, where W rows are output neurons.
// B-fragment == A-layout fragment of W (rows = N of tile, cols = K).
__device__ __forceinline__ v8f gemm_tile(const _Float16* A, int lda,
                                         const _Float16* Wt, int ldw, int K) {
    v8f c = {0.f, 0.f, 0.f, 0.f, 0.f, 0.f, 0.f, 0.f};
    for (int k0 = 0; k0 < K; k0 += 32) {
        v16h a = load_frag(A + k0, lda);
        v16h b = load_frag(Wt + k0, ldw);
        c = __builtin_amdgcn_wmma_f32_16x16x32_f16(false, a, false, b,
                                                   (short)0, c, false, false);
    }
    return c;
}

__device__ __forceinline__ float softplus_f(float x) {
    return (x > 20.f) ? x : log1pf(expf(x));
}

// Dense layer: out(16 x 128) = act( A(16 x K) @ W^T + bias ).
// Waves 0..7 each own one 16-wide output tile. actmode: 0=relu,1=softplus,2=identity
__device__ __forceinline__ void dense_layer(const _Float16* A, int K,
                                            const _Float16* Wg, const float* bias,
                                            _Float16* dst16, float* dst32, int actmode) {
    const int tid = threadIdx.x;
    const int wave = tid >> 5, lane = tid & 31;
    v8f c;
    if (wave < 8)
        c = gemm_tile(A, K, Wg + (size_t)(wave * 16) * K, K, K);
    __syncthreads();              // all reads of A done before anyone writes
    if (wave < 8) {
        int n0 = wave * 16;
        int col = lane & 15;
        int rbase = (lane & 16) ? 8 : 0;
        float bv = bias[n0 + col];
#pragma unroll
        for (int v = 0; v < 8; ++v) {
            float x = c[v] + bv;
            if (actmode == 0)      x = fmaxf(x, 0.f);
            else if (actmode == 1) x = softplus_f(x);
            int row = rbase + v;
            if (dst16) dst16[row * HH + n0 + col] = (_Float16)x;
            if (dst32) dst32[row * HH + n0 + col] = x;
        }
    }
    __syncthreads();
}

// Layer 3 + contraction: k[m,h] = dt_m * sum_d tanh( (h2 @ fW2^T)[m, h*32+d] + b2 ) * dv[m,d]
// Each of the 16 waves owns 8 h-values; per h, two 16x16 j-tiles (d0 = 0, 16).
__device__ __forceinline__ void layer3_contract(const _Float16* A /*16xH f16*/,
                                                const _Float16* fW2, const float* fb2,
                                                const float* dv /*16 x 32*/,
                                                float* kb /*16 x H*/, const float* dts) {
    const int tid = threadIdx.x;
    const int wave = tid >> 5, lane = tid & 31;
    const int col = lane & 15;
    const int rbase = (lane & 16) ? 8 : 0;
    for (int hh = 0; hh < 8; ++hh) {
        int h = wave * 8 + hh;
        // prefetch next h's weight tile while this one is in the WMMA pipe
        if (hh < 7)
            __builtin_prefetch((const void*)(fW2 + (size_t)((h + 1) * DD) * HH), 0, 1);
        float part[8];
#pragma unroll
        for (int v = 0; v < 8; ++v) part[v] = 0.f;
#pragma unroll
        for (int dblk = 0; dblk < 2; ++dblk) {
            int d0 = dblk * 16;
            int j0 = h * DD + d0;
            v8f c = gemm_tile(A, HH, fW2 + (size_t)j0 * HH, HH, HH);
            float bv = fb2[j0 + col];
#pragma unroll
            for (int v = 0; v < 8; ++v) {
                int row = rbase + v;
                float t = tanhf(c[v] + bv) * dv[row * DD + d0 + col];
                // butterfly-reduce across the 16-lane column group
                t += __shfl_xor(t, 1, 32);
                t += __shfl_xor(t, 2, 32);
                t += __shfl_xor(t, 4, 32);
                t += __shfl_xor(t, 8, 32);
                part[v] += t;
            }
        }
        if (col == 0) {
#pragma unroll
            for (int v = 0; v < 8; ++v) {
                int row = rbase + v;
                kb[row * HH + h] = part[v] * dts[row];
            }
        }
    }
}

// Issue one TDM 2D tile copy (rows x cols f16, row-major, contiguous) global -> LDS.
// D# layout per CDNA5 ISA.
// amdgpu-toolchain (clang-23) 6-arg builtin: (v4u g0, v8i g1, v4i g2, v4i g3, v8i g4, i32 cpol)
__device__ __forceinline__ void tdm_load_2d_f16(const _Float16* gsrc, unsigned int lds_byte_off,
                                                unsigned int rows, unsigned int cols) {
    unsigned long long ga = (unsigned long long)(uintptr_t)gsrc;
    v4u g0;
    g0[0] = 1u;                                           // count=1, user descriptor
    g0[1] = lds_byte_off;                                 // lds_addr (bytes)
    g0[2] = (unsigned int)(ga & 0xFFFFFFFFu);             // global_addr[31:0]
    g0[3] = (unsigned int)((ga >> 32) & 0x1FFFFFFu)       // global_addr[56:32]
          | (2u << 30);                                   // type = 2 ("image")
    v8i g1;
    g1[0] = (int)(1u << 16);                              // data_size = 1 -> 2 bytes
    g1[1] = (int)((cols & 0xFFFFu) << 16);                // tensor_dim0[15:0]
    g1[2] = (int)(((cols >> 16) & 0xFFFFu)                // tensor_dim0[31:16]
          | ((rows & 0xFFFFu) << 16));                    // tensor_dim1[15:0]
    g1[3] = (int)(((rows >> 16) & 0xFFFFu)                // tensor_dim1[31:16]
          | ((cols & 0xFFFFu) << 16));                    // tile_dim0
    g1[4] = (int)(rows & 0xFFFFu);                        // tile_dim1 (tile_dim2 = 0)
    g1[5] = (int)cols;                                    // tensor_dim0_stride[31:0]
    g1[6] = 0;                                            // stride0[47:32], stride1[15:0]
    g1[7] = 0;                                            // stride1[47:16]
    v4i gz4 = {0, 0, 0, 0};
    v8i gz8 = {0, 0, 0, 0, 0, 0, 0, 0};
    __builtin_amdgcn_tensor_load_to_lds(g0, g1, gz4, gz4, gz8, 0);
}

// ---------------- main Neural-CDE kernel ----------------
__global__ __launch_bounds__(NTHR)
void cde_kernel(const float* __restrict__ ts,
                const float* __restrict__ cd, const float* __restrict__ cc,
                const float* __restrict__ cb, const float* __restrict__ ca,
                const _Float16* __restrict__ iW0, const float* __restrict__ ib0,
                const _Float16* __restrict__ iW1, const float* __restrict__ ib1,
                const _Float16* __restrict__ iW2, const float* __restrict__ ib2,
                const _Float16* __restrict__ fW0, const float* __restrict__ fb0,
                const _Float16* __restrict__ fW1, const float* __restrict__ fb1,
                const _Float16* __restrict__ fW2, const float* __restrict__ fb2,
                const float* __restrict__ lW, const float* __restrict__ lb,
                float* __restrict__ out) {
    __shared__ float     yb[MT * HH];      // state y  (f32)
    __shared__ float     acc[MT * HH];     // RK weighted-k accumulator
    __shared__ float     kb[MT * HH];      // current k
    __shared__ float     dv[MT * DD];      // spline derivative per substep
    __shared__ _Float16  a16[MT * HH];     // ping
    __shared__ _Float16  b16[MT * HH];     // pong
    __shared__ float     dts[MT];
    __shared__ _Float16  wf0[WW * HH];     // f_W0 staged via TDM (32 KB)
    __shared__ _Float16  wf1[WW * WW];     // f_W1 staged via TDM (32 KB)

    const int tid = threadIdx.x;
    const int wave = tid >> 5, lane = tid & 31;
    const int bg = blockIdx.x * MT;

    // Stage the hot-loop hidden-layer weights into LDS with the Tensor Data Mover.
    // TDM ignores EXEC and is per-wave: issue from wave 0 only, then wait TENSORcnt.
    if (wave == 0) {
        tdm_load_2d_f16(fW0, (unsigned int)(uintptr_t)(void*)wf0, WW, HH);
        tdm_load_2d_f16(fW1, (unsigned int)(uintptr_t)(void*)wf1, WW, WW);
        __builtin_amdgcn_s_wait_tensorcnt(0);
    }

    // dt per batch row; initial control value a[0] -> a16 (16 x 32 f16)
    {
        int m = wave, d = lane;                 // NWAVE == MT, 32 lanes == DD
        if (d == 0) dts[m] = ts[(size_t)(bg + m) * TT + 1] - ts[(size_t)(bg + m) * TT];
        a16[m * HH + d] = (_Float16)ca[(size_t)(bg + m) * (TT - 1) * DD + d];
    }
    __syncthreads();

    // y0 = initial_mlp(a[0]): relu, relu, identity
    dense_layer(a16, DD, iW0, ib0, b16, nullptr, 0);
    dense_layer(b16, HH, iW1, ib1, a16, nullptr, 0);
    dense_layer(a16, HH, iW2, ib2, a16, yb, 2);   // y0 -> yb (f32) and a16 (f16)

    const float rkw[3] = {2.f / 9.f, 1.f / 3.f, 4.f / 9.f};
    const float alph[3] = {0.f, 0.5f, 0.75f};

    for (int s = 0; s < TT; ++s) {
        const int idx = (s < TT - 2) ? s : (TT - 2);
        for (int ss = 0; ss < 3; ++ss) {
            // spline derivative at t = ts[s] + alpha*dt (per batch row)
            {
                int m = wave, d = lane;
                size_t ci = ((size_t)(bg + m) * (TT - 1) + idx) * DD + d;
                float fr = ((float)(s - idx) + alph[ss]) * dts[m];
                dv[m * DD + d] = cb[ci] + fr * (2.f * cc[ci] + fr * 3.f * cd[ci]);
            }
            __syncthreads();

            // func_mlp hidden layers (softplus) — weights come from LDS (TDM-staged)
            dense_layer(a16, HH, wf0, fb0, b16, nullptr, 1);
            dense_layer(b16, HH, wf1, fb1, a16, nullptr, 1);
            // tanh output layer fused with deriv contraction -> kb (fW2 streamed from L2)
            layer3_contract(a16, fW2, fb2, dv, kb, dts);
            __syncthreads();

            // RK bookkeeping + next substep input (elementwise, 4 per thread)
#pragma unroll
            for (int j = 0; j < (MT * HH) / NTHR; ++j) {
                int i = tid + j * NTHR;
                int m = i >> 7;
                float kv = kb[i];
                if (ss == 0) {
                    acc[i] = rkw[0] * kv;
                    a16[i] = (_Float16)(yb[i] + 0.5f * kv);
                } else if (ss == 1) {
                    acc[i] += rkw[1] * kv;
                    a16[i] = (_Float16)(yb[i] + 0.75f * kv);
                } else {
                    float yn = yb[i] + dts[m] * (acc[i] + rkw[2] * kv);
                    yb[i] = yn;
                    a16[i] = (_Float16)yn;      // input for next step's k1
                }
            }
            __syncthreads();
        }

        // out[b][s] = sigmoid(y . l_W + l_b); one wave per batch row
        {
            int m = wave;
            float sum = yb[m * HH + lane]       * lW[lane]
                      + yb[m * HH + lane + 32]  * lW[lane + 32]
                      + yb[m * HH + lane + 64]  * lW[lane + 64]
                      + yb[m * HH + lane + 96]  * lW[lane + 96];
            sum += __shfl_xor(sum, 1, 32);
            sum += __shfl_xor(sum, 2, 32);
            sum += __shfl_xor(sum, 4, 32);
            sum += __shfl_xor(sum, 8, 32);
            sum += __shfl_xor(sum, 16, 32);
            if (lane == 0)
                out[(size_t)(bg + m) * TT + s] = 1.f / (1.f + expf(-(sum + lb[0])));
        }
        __syncthreads();
    }
}

// ---------------- host launch ----------------
extern "C" void kernel_launch(void* const* d_in, const int* in_sizes, int n_in,
                              void* d_out, int out_size, void* d_ws, size_t ws_size,
                              hipStream_t stream) {
    const float* ts  = (const float*)d_in[0];
    const float* cd  = (const float*)d_in[1];
    const float* cc  = (const float*)d_in[2];
    const float* cb  = (const float*)d_in[3];
    const float* ca  = (const float*)d_in[4];
    const float* iW0 = (const float*)d_in[5];   const float* ib0 = (const float*)d_in[6];
    const float* iW1 = (const float*)d_in[7];   const float* ib1 = (const float*)d_in[8];
    const float* iW2 = (const float*)d_in[9];   const float* ib2 = (const float*)d_in[10];
    const float* fW0 = (const float*)d_in[11];  const float* fb0 = (const float*)d_in[12];
    const float* fW1 = (const float*)d_in[13];  const float* fb1 = (const float*)d_in[14];
    const float* fW2 = (const float*)d_in[15];  const float* fb2 = (const float*)d_in[16];
    const float* lW  = (const float*)d_in[17];  const float* lb  = (const float*)d_in[18];

    // f16 weight workspace layout (element counts)
    const int n_iW0 = WW * DD;        // 4096
    const int n_iW1 = WW * WW;        // 16384
    const int n_iW2 = HH * WW;        // 16384
    const int n_fW0 = WW * HH;        // 16384
    const int n_fW1 = WW * WW;        // 16384
    const int n_fW2 = HH * DD * WW;   // 524288

    _Float16* w = (_Float16*)d_ws;
    _Float16* iW0h = w;               w += n_iW0;
    _Float16* iW1h = w;               w += n_iW1;
    _Float16* iW2h = w;               w += n_iW2;
    _Float16* fW0h = w;               w += n_fW0;
    _Float16* fW1h = w;               w += n_fW1;
    _Float16* fW2h = w;               w += n_fW2;

    auto cvt = [&](const float* s, _Float16* d, int n) {
        cvt_f32_f16<<<(n + 255) / 256, 256, 0, stream>>>(s, d, n);
    };
    cvt(iW0, iW0h, n_iW0);
    cvt(iW1, iW1h, n_iW1);
    cvt(iW2, iW2h, n_iW2);
    cvt(fW0, fW0h, n_fW0);
    cvt(fW1, fW1h, n_fW1);
    cvt(fW2, fW2h, n_fW2);

    cde_kernel<<<BB / MT, NTHR, 0, stream>>>(
        ts, cd, cc, cb, ca,
        iW0h, ib0, iW1h, ib1, iW2h, ib2,
        fW0h, fb0, fW1h, fb1, fW2h, fb2,
        lW, lb, (float*)d_out);
}
